// GraphLSTMVAE_41712722379112
// MI455X (gfx1250) — matrix operand order repaired and checked by reading, not compiled, not run on hardware
//
#include <hip/hip_runtime.h>
#include <math.h>

// ---------------- problem constants (match reference) ----------------
#define H      128
#define MSGS   512001           // M
#define MPAD   512128           // M padded to multiple of 128
#define NNODES 204800           // N = B*L
#define NSEQ   512              // B
#define LSEQ   400              // L
#define HH     64               // H/2

typedef __attribute__((ext_vector_type(16))) __bf16 v16bf;
typedef __attribute__((ext_vector_type(2)))  __bf16 v2bf;
typedef __attribute__((ext_vector_type(8)))  float  v8f;

union FragB { v16bf v; unsigned u[8]; };

// ---------------- bf16 helpers ----------------
__device__ __forceinline__ unsigned short f2bf(float f) {
  union { float f; unsigned u; } x; x.f = f;
  unsigned r = x.u + 0x7FFFu + ((x.u >> 16) & 1u);   // round-to-nearest-even
  return (unsigned short)(r >> 16);
}
__device__ __forceinline__ float bf2f(unsigned short b) {
  union { unsigned u; float f; } x; x.u = ((unsigned)b) << 16; return x.f;
}
// packed f32 -> bf16x2 convert (RNE). Builtin if present, else CDNA5
// V_CVT_PK_BF16_F32 via inline asm (one VOP instead of ~6 VALU ops).
__device__ __forceinline__ unsigned pack2(float lo, float hi) {
#if __has_builtin(__builtin_amdgcn_cvt_pk_bf16_f32)
  union { v2bf v; unsigned u; } p;
  p.v = __builtin_amdgcn_cvt_pk_bf16_f32(lo, hi);
  return p.u;
#else
  unsigned d;
  asm volatile("v_cvt_pk_bf16_f32 %0, %1, %2" : "=v"(d) : "v"(lo), "v"(hi));
  return d;
#endif
}
// packed bf16 add (VOP3P V_PK_ADD_BF16) — no clang builtin, use CDNA5 asm
__device__ __forceinline__ unsigned pk_add_bf16(unsigned a, unsigned b) {
  unsigned d;
  asm volatile("v_pk_add_bf16 %0, %1, %2" : "=v"(d) : "v"(a), "v"(b));
  return d;
}
__device__ __forceinline__ float sigm(float x) { return 1.0f / (1.0f + __expf(-x)); }

// ---------------- WMMA wrapper (CDNA5 bf16, f32 accumulate) ----------------
__device__ __forceinline__ v8f wmma_bf16(v16bf a, v16bf b, v8f c) {
  return __builtin_amdgcn_wmma_f32_16x16x32_bf16(false, a, false, b, (short)0, c,
                                                 false, false);
}

// A-matrix (16x32 bf16) per-lane K offset for VGPR v (ISA 7.12.2):
// lanes 0-15: V0-3 -> K 0..7, V4-7 -> K 16..23 ; lanes 16-31: +8
__device__ __forceinline__ int a_koff(int v, int lane) {
  return ((v < 4) ? 2 * v : 2 * v + 8) + ((lane & 16) ? 8 : 0);
}
// B-matrix (32x16 bf16): lane = col, same K order as A:
// lanes 0-15: V0-3 -> K 0..7, V4-7 -> K 16..23 ; lanes 16-31: +8
__device__ __forceinline__ int b_koff(int v, int lane) {
  return ((v < 4) ? 2 * v : 2 * v + 8) + ((lane & 16) ? 8 : 0);
}

// ---------------- K0: f32 -> bf16 weight conversion ----------------
__global__ void cvt_kernel(const float* __restrict__ s, unsigned short* __restrict__ d, int n) {
  int i = blockIdx.x * 256 + threadIdx.x;
  if (i < n) d[i] = f2bf(s[i]);
}

// ---------------- K1: local = f_bond @ w_local.T ; messages0 = relu(local) ----
__global__ __launch_bounds__(256) void local_init_kernel(
    const float* __restrict__ f_bond, const float* __restrict__ w_local,
    unsigned short* __restrict__ localb, unsigned short* __restrict__ msg0) {
  long idx = (long)blockIdx.x * 256 + threadIdx.x;   // over MPAD*H
  long m = idx >> 7; int h = (int)(idx & 127);
  if (m >= MPAD) return;
  float s = 0.0f;
  if (m < MSGS) {
#pragma unroll
    for (int k = 0; k < 8; ++k) s += f_bond[m * 8 + k] * w_local[h * 8 + k];
  }
  localb[idx] = f2bf(s);
  msg0[idx]   = f2bf(fmaxf(s, 0.0f));
}

// ---------------- K2: one fused MPN+GRU iteration ----------------
// messages_out = GRU( relu(local + (gather3(messages_in) @ w_msg.T)), messages_in )
// One wave per 16-row tile; 4 waves per block. All row-indexed buffers are
// MPAD rows and fully initialized, so loads are unguarded (no exec-mask
// branches); only the output store masks dummy/pad rows.
__global__ __launch_bounds__(128) void mpn_iter_kernel(
    const unsigned short* __restrict__ msg_in, unsigned short* __restrict__ msg_out,
    const unsigned short* __restrict__ localb, const int* __restrict__ mgraph,
    const unsigned short* __restrict__ wmsg,   // [128][128] bf16
    const unsigned short* __restrict__ wih,    // [384][128] bf16
    const unsigned short* __restrict__ whh,    // [384][128] bf16
    const float* __restrict__ bih, const float* __restrict__ bhh) {
  __shared__ unsigned short nm[4][16][136];          // new_msg tiles (padded stride)
  const int tid = threadIdx.x;
  const int wv = tid >> 5, lane = tid & 31;
  const long r0 = ((long)blockIdx.x * 4 + wv) * 16;  // tile row base in M
  const int mrow = lane & 15;
  const long gm = r0 + mrow;
  // clamp gather-index row so the mgraph load is always in-bounds (result of
  // clamped rows is masked to zero at the final store)
  const long gmc = (gm < (long)MSGS) ? gm : (long)(MSGS - 1);
  const int n0 = mgraph[gmc * 3 + 0];
  const int n1 = mgraph[gmc * 3 + 1];
  const int n2 = mgraph[gmc * 3 + 2];

  // --- gather+sum neighbors straight into A fragments (K=128 -> 4 tiles) ---
  // packed bf16 adds: 2 VALU ops per dword instead of unpack/add/repack
  FragB agg[4];
#pragma unroll
  for (int kt = 0; kt < 4; ++kt) {
#pragma unroll
    for (int v = 0; v < 8; ++v) {
      int k = kt * 32 + a_koff(v, lane);
      unsigned a0 = *(const unsigned*)(msg_in + (long)n0 * H + k);
      unsigned a1 = *(const unsigned*)(msg_in + (long)n1 * H + k);
      unsigned a2 = *(const unsigned*)(msg_in + (long)n2 * H + k);
      agg[kt].u[v] = pk_add_bf16(pk_add_bf16(a0, a1), a2);
    }
  }

  // --- GEMM1: new_msg = relu(local + agg @ wmsg.T), staged to LDS ---
#pragma unroll 1
  for (int ct = 0; ct < 8; ++ct) {
    v8f acc = {0.f, 0.f, 0.f, 0.f, 0.f, 0.f, 0.f, 0.f};
    const int n = ct * 16 + (lane & 15);
#pragma unroll
    for (int kt = 0; kt < 4; ++kt) {
      FragB b;
#pragma unroll
      for (int v = 0; v < 8; ++v)
        b.u[v] = *(const unsigned*)(wmsg + (long)n * H + kt * 32 + b_koff(v, lane));
      acc = wmma_bf16(agg[kt].v, b.v, acc);
    }
    // pairwise epilogue: v_max relu + packed bf16 convert per element pair
#pragma unroll
    for (int e = 0; e < 8; e += 2) {
      int rr = e + ((lane & 16) ? 8 : 0);
      long g0 = r0 + rr;                         // < MPAD always; pad rows are 0
      float v0 = fmaxf(acc[e]     + bf2f(localb[g0 * H + n]),       0.0f);
      float v1 = fmaxf(acc[e + 1] + bf2f(localb[(g0 + 1) * H + n]), 0.0f);
      unsigned p = pack2(v0, v1);
      nm[wv][rr][n]     = (unsigned short)p;
      nm[wv][rr + 1][n] = (unsigned short)(p >> 16);
    }
  }
  __syncthreads();

  // --- load A fragments: new_msg from LDS, old messages from global ---
  FragB nma[4], oldm[4];
#pragma unroll
  for (int kt = 0; kt < 4; ++kt) {
#pragma unroll
    for (int v = 0; v < 8; ++v) {
      int k = kt * 32 + a_koff(v, lane);
      nma[kt].u[v]  = *(const unsigned*)(&nm[wv][mrow][k]);
      oldm[kt].u[v] = *(const unsigned*)(msg_in + gm * H + k);   // pad rows are 0
    }
  }

  // --- GEMM2/3 (GRU gates) + elementwise update, per 16-col tile ---
#pragma unroll 1
  for (int ct = 0; ct < 8; ++ct) {
    v8f air = {0.f,0.f,0.f,0.f,0.f,0.f,0.f,0.f}, aiz = air, ain = air;
    v8f ahr = air, ahz = air, ahn = air;
    const int n = ct * 16 + (lane & 15);
#pragma unroll
    for (int kt = 0; kt < 4; ++kt) {
      FragB br, bz, bn, cr, cz, cn;
#pragma unroll
      for (int v = 0; v < 8; ++v) {
        int k = kt * 32 + b_koff(v, lane);
        br.u[v] = *(const unsigned*)(wih + (long)(n)       * H + k);
        bz.u[v] = *(const unsigned*)(wih + (long)(128 + n) * H + k);
        bn.u[v] = *(const unsigned*)(wih + (long)(256 + n) * H + k);
        cr.u[v] = *(const unsigned*)(whh + (long)(n)       * H + k);
        cz.u[v] = *(const unsigned*)(whh + (long)(128 + n) * H + k);
        cn.u[v] = *(const unsigned*)(whh + (long)(256 + n) * H + k);
      }
      air = wmma_bf16(nma[kt].v,  br.v, air);
      aiz = wmma_bf16(nma[kt].v,  bz.v, aiz);
      ain = wmma_bf16(nma[kt].v,  bn.v, ain);
      ahr = wmma_bf16(oldm[kt].v, cr.v, ahr);
      ahz = wmma_bf16(oldm[kt].v, cz.v, ahz);
      ahn = wmma_bf16(oldm[kt].v, cn.v, ahn);
    }
    const float bir = bih[n], biz = bih[128 + n], bin = bih[256 + n];
    const float bhr = bhh[n], bhz = bhh[128 + n], bhn = bhh[256 + n];
#pragma unroll
    for (int e = 0; e < 8; e += 2) {
      int rr = e + ((lane & 16) ? 8 : 0);
      long g0 = r0 + rr, g1 = g0 + 1;
      float old0 = bf2f(msg_in[g0 * H + n]);     // unguarded: pad rows are 0
      float old1 = bf2f(msg_in[g1 * H + n]);
      float r0g = sigm(air[e]     + bir + ahr[e]     + bhr);
      float r1g = sigm(air[e + 1] + bir + ahr[e + 1] + bhr);
      float z0  = sigm(aiz[e]     + biz + ahz[e]     + bhz);
      float z1  = sigm(aiz[e + 1] + biz + ahz[e + 1] + bhz);
      float n0g = tanhf(ain[e]     + bin + r0g * (ahn[e]     + bhn));
      float n1g = tanhf(ain[e + 1] + bin + r1g * (ahn[e + 1] + bhn));
      float val0 = (1.0f - z0) * n0g + z0 * old0;
      float val1 = (1.0f - z1) * n1g + z1 * old1;
      if (g0 == 0 || g0 >= MSGS) val0 = 0.0f;    // dummy row + padding stay 0
      if (g1 >= MSGS) val1 = 0.0f;
      unsigned p = pack2(val0, val1);
      msg_out[g0 * H + n] = (unsigned short)p;
      msg_out[g1 * H + n] = (unsigned short)(p >> 16);
    }
  }
}

// ---------------- K3: node embedding = relu([f_nuc, gather3(msgs)] @ Wne.T) ----
__global__ __launch_bounds__(256) void node_embed_kernel(
    const unsigned short* __restrict__ msgs, const int* __restrict__ ngraph,
    const float* __restrict__ f_nuc, const float* __restrict__ wne, /* [128][132] f32 */
    unsigned short* __restrict__ nuc_emb) {
  __shared__ unsigned short wl[128 * 132];
  __shared__ float nb[2][128];
  const int tid = threadIdx.x;
  for (int i = tid; i < 128 * 132; i += 256) wl[i] = f2bf(wne[i]);
  __syncthreads();
  const long nbase = (long)blockIdx.x * 64;
  const int sub = tid >> 7, col = tid & 127;
  for (int g = 0; g < 32; ++g) {
    long node = nbase + g * 2 + sub;
    float s = 0.0f;
    {
      long b = node * 3;
      int i0 = ngraph[b], i1 = ngraph[b + 1], i2 = ngraph[b + 2];
      s = bf2f(msgs[(long)i0 * H + col]) + bf2f(msgs[(long)i1 * H + col]) +
          bf2f(msgs[(long)i2 * H + col]);
    }
    nb[sub][col] = s;
    __syncthreads();
    {
      const unsigned short* wr = wl + col * 132;
      float acc = 0.0f;
#pragma unroll
      for (int k = 0; k < 4; ++k) acc += f_nuc[node * 4 + k] * bf2f(wr[k]);
      const float* nbr = nb[sub];
#pragma unroll 8
      for (int k = 0; k < 128; ++k) acc += nbr[k] * bf2f(wr[4 + k]);
      nuc_emb[node * H + col] = f2bf(fmaxf(acc, 0.0f));
    }
    __syncthreads();
  }
}

// ---------------- K4: one-direction LSTM + running channel max ----------------
// 16 sequences per block; wave w owns a 16-wide c-slice of all four gates.
__global__ __launch_bounds__(128) void lstm_dir_kernel(
    const unsigned short* __restrict__ x,    // nuc_emb [N][128] bf16
    const unsigned short* __restrict__ wih,  // [256][128] bf16
    const unsigned short* __restrict__ whh,  // [256][64]  bf16
    const float* __restrict__ bih, const float* __restrict__ bhh,
    float* __restrict__ out, int dir) {
  __shared__ unsigned short hsh[16][68];     // padded h state tile
  const int tid = threadIdx.x, wv = tid >> 5, lane = tid & 31;
  const int seqBase = blockIdx.x * 16;
  const int cs = wv * 16;
  const int nloc = lane & 15;
  for (int i = tid; i < 16 * 68; i += 128) ((unsigned short*)hsh)[i] = 0;
  v8f c = {0.f,0.f,0.f,0.f,0.f,0.f,0.f,0.f};
  v8f hmax = {-1e30f,-1e30f,-1e30f,-1e30f,-1e30f,-1e30f,-1e30f,-1e30f};
  const float bi = bih[       cs + nloc] + bhh[       cs + nloc];
  const float bf = bih[ 64  + cs + nloc] + bhh[ 64  + cs + nloc];
  const float bg = bih[128  + cs + nloc] + bhh[128  + cs + nloc];
  const float bo = bih[192  + cs + nloc] + bhh[192  + cs + nloc];
  __syncthreads();

  for (int t = 0; t < LSEQ; ++t) {
    const int tt = dir ? (LSEQ - 1 - t) : t;
    // x A-fragments (rows stride L apart: one row per sequence)
    FragB xa[4];
    const unsigned short* xr = x + ((long)(seqBase + nloc) * LSEQ + tt) * H;
#pragma unroll
    for (int kt = 0; kt < 4; ++kt)
#pragma unroll
      for (int v = 0; v < 8; ++v)
        xa[kt].u[v] = *(const unsigned*)(xr + kt * 32 + a_koff(v, lane));
    // h A-fragments from LDS (K=64)
    FragB ha[2];
#pragma unroll
    for (int kt = 0; kt < 2; ++kt)
#pragma unroll
      for (int v = 0; v < 8; ++v)
        ha[kt].u[v] = *(const unsigned*)(&hsh[nloc][kt * 32 + a_koff(v, lane)]);
    __syncthreads();   // all waves done reading h before anyone rewrites it

    v8f gi = {0.f,0.f,0.f,0.f,0.f,0.f,0.f,0.f}, gf = gi, gg = gi, go = gi;
#pragma unroll
    for (int kt = 0; kt < 4; ++kt) {
      FragB b0, b1, b2, b3;
#pragma unroll
      for (int v = 0; v < 8; ++v) {
        int k = kt * 32 + b_koff(v, lane);
        b0.u[v] = *(const unsigned*)(wih + (long)(      cs + nloc) * H + k);
        b1.u[v] = *(const unsigned*)(wih + (long)( 64 + cs + nloc) * H + k);
        b2.u[v] = *(const unsigned*)(wih + (long)(128 + cs + nloc) * H + k);
        b3.u[v] = *(const unsigned*)(wih + (long)(192 + cs + nloc) * H + k);
      }
      gi = wmma_bf16(xa[kt].v, b0.v, gi);
      gf = wmma_bf16(xa[kt].v, b1.v, gf);
      gg = wmma_bf16(xa[kt].v, b2.v, gg);
      go = wmma_bf16(xa[kt].v, b3.v, go);
    }
#pragma unroll
    for (int kt = 0; kt < 2; ++kt) {
      FragB b0, b1, b2, b3;
#pragma unroll
      for (int v = 0; v < 8; ++v) {
        int k = kt * 32 + b_koff(v, lane);
        b0.u[v] = *(const unsigned*)(whh + (long)(      cs + nloc) * HH + k);
        b1.u[v] = *(const unsigned*)(whh + (long)( 64 + cs + nloc) * HH + k);
        b2.u[v] = *(const unsigned*)(whh + (long)(128 + cs + nloc) * HH + k);
        b3.u[v] = *(const unsigned*)(whh + (long)(192 + cs + nloc) * HH + k);
      }
      gi = wmma_bf16(ha[kt].v, b0.v, gi);
      gf = wmma_bf16(ha[kt].v, b1.v, gf);
      gg = wmma_bf16(ha[kt].v, b2.v, gg);
      go = wmma_bf16(ha[kt].v, b3.v, go);
    }
#pragma unroll
    for (int e = 0; e < 8; e += 2) {
      float i0 = sigm(gi[e] + bi),     f0 = sigm(gf[e] + bf);
      float g0 = tanhf(gg[e] + bg),    o0 = sigm(go[e] + bo);
      float i1 = sigm(gi[e + 1] + bi), f1 = sigm(gf[e + 1] + bf);
      float g1 = tanhf(gg[e + 1] + bg), o1 = sigm(go[e + 1] + bo);
      c[e]     = f0 * c[e]     + i0 * g0;
      c[e + 1] = f1 * c[e + 1] + i1 * g1;
      float h0 = o0 * tanhf(c[e]);
      float h1 = o1 * tanhf(c[e + 1]);
      hmax[e]     = fmaxf(hmax[e], h0);
      hmax[e + 1] = fmaxf(hmax[e + 1], h1);
      int rr = e + ((lane & 16) ? 8 : 0);
      unsigned p = pack2(h0, h1);
      hsh[rr][cs + nloc]     = (unsigned short)p;
      hsh[rr + 1][cs + nloc] = (unsigned short)(p >> 16);
    }
    __syncthreads();   // new h visible to all waves for next step
  }
#pragma unroll
  for (int e = 0; e < 8; ++e) {
    int rr = e + ((lane & 16) ? 8 : 0);
    out[(long)(seqBase + rr) * H + dir * HH + cs + nloc] = hmax[e];
  }
}

// ---------------- host-side launch ----------------
extern "C" void kernel_launch(void* const* d_in, const int* in_sizes, int n_in,
                              void* d_out, int out_size, void* d_ws, size_t ws_size,
                              hipStream_t stream) {
  (void)in_sizes; (void)n_in; (void)out_size; (void)ws_size;
  const float* f_nuc      = (const float*)d_in[0];
  const float* f_bond     = (const float*)d_in[1];
  const int*   node_graph = (const int*)d_in[2];
  const int*   msg_graph  = (const int*)d_in[3];
  const float* w_local    = (const float*)d_in[5];
  const float* w_msg      = (const float*)d_in[6];
  const float* w_node_emb = (const float*)d_in[7];
  const float* gru_w_ih   = (const float*)d_in[8];
  const float* gru_w_hh   = (const float*)d_in[9];
  const float* gru_b_ih   = (const float*)d_in[10];
  const float* gru_b_hh   = (const float*)d_in[11];
  const float* lw_ih_f    = (const float*)d_in[12];
  const float* lw_hh_f    = (const float*)d_in[13];
  const float* lb_ih_f    = (const float*)d_in[14];
  const float* lb_hh_f    = (const float*)d_in[15];
  const float* lw_ih_b    = (const float*)d_in[16];
  const float* lw_hh_b    = (const float*)d_in[17];
  const float* lb_ih_b    = (const float*)d_in[18];
  const float* lb_hh_b    = (const float*)d_in[19];
  float* out = (float*)d_out;

  // workspace carve-up (bf16 buffers; ~446 MB total)
  char* ws = (char*)d_ws;
  size_t off = 0;
  auto carve = [&](size_t bytes) { size_t o = off; off += (bytes + 255) & ~(size_t)255; return o; };
  const size_t msgBytes = (size_t)MPAD * H * 2;
  unsigned short* msgA    = (unsigned short*)(ws + carve(msgBytes));
  unsigned short* msgB    = (unsigned short*)(ws + carve(msgBytes));
  unsigned short* localb  = (unsigned short*)(ws + carve(msgBytes));
  unsigned short* nuc_emb = (unsigned short*)(ws + carve((size_t)NNODES * H * 2));
  unsigned short* wmsg_b  = (unsigned short*)(ws + carve(128 * 128 * 2));
  unsigned short* wih_b   = (unsigned short*)(ws + carve(384 * 128 * 2));
  unsigned short* whh_b   = (unsigned short*)(ws + carve(384 * 128 * 2));
  unsigned short* wihF    = (unsigned short*)(ws + carve(256 * 128 * 2));
  unsigned short* whhF    = (unsigned short*)(ws + carve(256 * 64 * 2));
  unsigned short* wihB    = (unsigned short*)(ws + carve(256 * 128 * 2));
  unsigned short* whhB    = (unsigned short*)(ws + carve(256 * 64 * 2));

  auto cv = [&](const float* s, unsigned short* d, int n) {
    cvt_kernel<<<dim3((n + 255) / 256), dim3(256), 0, stream>>>(s, d, n);
  };
  cv(w_msg,    wmsg_b, 128 * 128);
  cv(gru_w_ih, wih_b,  384 * 128);
  cv(gru_w_hh, whh_b,  384 * 128);
  cv(lw_ih_f,  wihF,   256 * 128);
  cv(lw_hh_f,  whhF,   256 * 64);
  cv(lw_ih_b,  wihB,   256 * 128);
  cv(lw_hh_b,  whhB,   256 * 64);

  // local potentials + initial messages
  local_init_kernel<<<dim3((unsigned)(((size_t)MPAD * H) / 256)), dim3(256), 0, stream>>>(
      f_bond, w_local, localb, msgA);

  // 4 fused MPN+GRU iterations, ping-pong (ends in msgA)
  const unsigned mpnBlocks = MPAD / 64;  // 4 wave-tiles of 16 rows per block
  unsigned short* pin = msgA; unsigned short* pout = msgB;
  for (int it = 0; it < 4; ++it) {
    mpn_iter_kernel<<<dim3(mpnBlocks), dim3(128), 0, stream>>>(
        pin, pout, localb, msg_graph, wmsg_b, wih_b, whh_b, gru_b_ih, gru_b_hh);
    unsigned short* t = pin; pin = pout; pout = t;
  }

  // node embedding
  node_embed_kernel<<<dim3(NNODES / 64), dim3(256), 0, stream>>>(
      pin, node_graph, f_nuc, w_node_emb, nuc_emb);

  // bidirectional LSTM with fused running max-pool
  lstm_dir_kernel<<<dim3(NSEQ / 16), dim3(128), 0, stream>>>(
      nuc_emb, wihF, whhF, lb_ih_f, lb_hh_f, out, 0);
  lstm_dir_kernel<<<dim3(NSEQ / 16), dim3(128), 0, stream>>>(
      nuc_emb, wihB, whhB, lb_ih_b, lb_hh_b, out, 1);
}
